// MoEModel_16312285790340
// MI455X (gfx1250) — matrix operand order repaired and checked by reading, not compiled, not run on hardware
//
#include <hip/hip_runtime.h>
#include <hip/hip_bf16.h>
#include <math.h>

// Problem constants (match reference)
constexpr int Bc = 4, Sc = 2048, Hc = 1024, Ec = 8, Fc = 4096, KT = 2;
constexpr int NT = Bc * Sc;          // 8192 tokens
constexpr int TM = 32;               // tokens per FFN block (LDS tile rows)

typedef __attribute__((ext_vector_type(16))) __bf16 bf16x16;
typedef __attribute__((ext_vector_type(8)))  __bf16 bf16x8;
typedef __attribute__((ext_vector_type(8)))  float  floatx8;

union AFrag { bf16x16 v; bf16x8 h[2]; };

// ---------------------------------------------------------------------------
// 0) Zero output + expert counters (d_out/d_ws are poisoned by the harness)
// ---------------------------------------------------------------------------
__global__ void moe_init_kernel(float* __restrict__ out, int* __restrict__ cnt, int n) {
    int i = blockIdx.x * blockDim.x + threadIdx.x;
    if (i < n) out[i] = 0.0f;
    if (i < Ec) cnt[i] = 0;
}

// ---------------------------------------------------------------------------
// 1) x (fp32) -> xb (bf16)
// ---------------------------------------------------------------------------
__global__ void moe_cvt_x_kernel(const float* __restrict__ x, __bf16* __restrict__ xb, int n) {
    int i = blockIdx.x * blockDim.x + threadIdx.x;
    if (i < n) xb[i] = (__bf16)x[i];
}

// ---------------------------------------------------------------------------
// 2) Per-expert transpose + fp32->bf16: in [E][R][C] -> out [E][C][R]
//    (gives WMMA B-fragments contiguous-in-K rows)
// ---------------------------------------------------------------------------
__global__ void moe_transpose_cvt_kernel(const float* __restrict__ in,
                                         __bf16* __restrict__ out, int R, int C) {
    __shared__ float tile[32][33];
    const int e  = blockIdx.z;
    const float* inp  = in  + (size_t)e * R * C;
    __bf16*      outp = out + (size_t)e * R * C;
    const int c0 = blockIdx.x * 32, r0 = blockIdx.y * 32;
    const int tx = threadIdx.x, ty = threadIdx.y;   // block (32, 8)
#pragma unroll
    for (int i = 0; i < 4; ++i) {
        int r = r0 + ty + i * 8;
        tile[ty + i * 8][tx] = inp[(size_t)r * C + c0 + tx];
    }
    __syncthreads();
#pragma unroll
    for (int i = 0; i < 4; ++i) {
        int c = c0 + ty + i * 8;
        outp[(size_t)c * R + r0 + tx] = (__bf16)tile[tx][ty + i * 8];
    }
}

// ---------------------------------------------------------------------------
// 3) Gating: logits = x @ Wg^T, softmax, top-2; build compacted expert lists.
//    One wave (32 lanes) per token.
// ---------------------------------------------------------------------------
__global__ void moe_gating_kernel(const float* __restrict__ x, const float* __restrict__ wg,
                                  float* __restrict__ topw, int* __restrict__ toklist,
                                  int* __restrict__ cnt) {
    const int wave = threadIdx.x >> 5;
    const int lane = threadIdx.x & 31;
    const int t = blockIdx.x * 8 + wave;
    if (t >= NT) return;
    const float* xr = x + (size_t)t * Hc;
    float acc[Ec];
#pragma unroll
    for (int e = 0; e < Ec; ++e) acc[e] = 0.0f;
    for (int h = lane; h < Hc; h += 32) {
        float xv = xr[h];
#pragma unroll
        for (int e = 0; e < Ec; ++e) acc[e] += xv * wg[e * Hc + h];
    }
#pragma unroll
    for (int e = 0; e < Ec; ++e)
        for (int off = 16; off > 0; off >>= 1) acc[e] += __shfl_xor(acc[e], off, 32);
    if (lane == 0) {
        float mx = acc[0];
#pragma unroll
        for (int e = 1; e < Ec; ++e) mx = fmaxf(mx, acc[e]);
        float p[Ec], s = 0.0f;
#pragma unroll
        for (int e = 0; e < Ec; ++e) { p[e] = __expf(acc[e] - mx); s += p[e]; }
        float inv = 1.0f / s;
#pragma unroll
        for (int e = 0; e < Ec; ++e) p[e] *= inv;
        int e0 = 0;
#pragma unroll
        for (int e = 1; e < Ec; ++e) if (p[e] > p[e0]) e0 = e;
        int e1 = (e0 == 0) ? 1 : 0;
#pragma unroll
        for (int e = 0; e < Ec; ++e) if (e != e0 && p[e] > p[e1]) e1 = e;
        topw[t * 2 + 0] = p[e0];
        topw[t * 2 + 1] = p[e1];
        int pos0 = atomicAdd(&cnt[e0], 1);
        toklist[e0 * NT + pos0] = (t << 1);
        int pos1 = atomicAdd(&cnt[e1], 1);
        toklist[e1 * NT + pos1] = (t << 1) | 1;
    }
}

__device__ __forceinline__ float gelu_erf(float v) {
    return 0.5f * v * (1.0f + erff(v * 0.70710678118654752f));
}

// ---------------------------------------------------------------------------
// 4) Fused expert FFN over routed tokens:
//    stage A: h = gelu(x @ w1 + b1)   (32 x 4096, bf16 LDS tile, 256 KB)
//    stage B: out += gate * (h @ w2 + b2)
//    Block: 256 threads = 8 waves. Wave (mi, wn): mi picks 16-row half,
//    wn picks a column range. 4 accumulators -> 4 WMMA per K-step.
// ---------------------------------------------------------------------------
__global__ __launch_bounds__(256) void moe_ffn_kernel(
    const __bf16* __restrict__ xb,      // [NT][H]
    const __bf16* __restrict__ w1t,     // [E][F][H]  (row n contiguous in K=H)
    const float*  __restrict__ b1,      // [E][F]
    const __bf16* __restrict__ w2t,     // [E][H][F]  (row n contiguous in K=F)
    const float*  __restrict__ b2,      // [E][H]
    const float*  __restrict__ topw,    // [NT*2]
    const int*    __restrict__ toklist, // [E][NT]
    const int*    __restrict__ cnt,     // [E]
    float*        __restrict__ out)     // [NT][H]
{
    __shared__ __bf16 hbuf[TM * Fc];    // 256 KB of the 320 KB WGP LDS
    __shared__ int    s_tok[TM];
    __shared__ float  s_gw[TM];
    __shared__ int    s_val[TM];

    const int e = blockIdx.y;
    const int c = cnt[e];
    const int tile = blockIdx.x;
    if (tile * TM >= c) return;         // uniform early exit

    const int tid = threadIdx.x;
    if (tid < TM) {
        int idx = tile * TM + tid;
        if (idx < c) {
            int en = toklist[e * NT + idx];
            s_tok[tid] = en >> 1;
            s_gw[tid]  = topw[en];
            s_val[tid] = 1;
        } else {
            s_tok[tid] = 0; s_gw[tid] = 0.0f; s_val[tid] = 0;
        }
    }
    __syncthreads();

    const int wave  = tid >> 5;
    const int lane  = tid & 31;
    const int mi    = wave & 1;         // which 16-row half of the 32-token tile
    const int wn    = wave >> 1;        // column-range owner (0..3)
    const int lhalf = lane >> 4;        // 0 or 1
    const int lmod  = lane & 15;
    const int mbase = mi * 16;

    // -------- stage A: h = gelu(x @ w1 + b1) --------
    const __bf16* arow = xb + (size_t)s_tok[mbase + lmod] * Hc; // A row for this lane
    const __bf16* w1e  = w1t + (size_t)e * Fc * Hc;
    const float*  b1e  = b1 + e * Fc;

    for (int g = 0; g < 16; ++g) {                 // 16 groups of 64 columns
        const int n0 = wn * 1024 + g * 64;
        floatx8 acc[4] = {};
        for (int k0 = 0; k0 < Hc; k0 += 32) {
            AFrag a;
            const __bf16* ap = arow + k0 + (lhalf << 4);
            a.h[0] = *(const bf16x8*)(ap);
            a.h[1] = *(const bf16x8*)(ap + 8);
#pragma unroll
            for (int j = 0; j < 4; ++j) {
                const int n = n0 + j * 16 + lmod;
                const __bf16* bp = w1e + (size_t)n * Hc + k0 + (lhalf << 4);
                bf16x16 b = *(const bf16x16*)bp;
                acc[j] = __builtin_amdgcn_wmma_f32_16x16x32_bf16(
                    false, a.v, false, b, (short)0, acc[j], false, false);
            }
        }
#pragma unroll
        for (int j = 0; j < 4; ++j) {
            const int n = n0 + j * 16 + lmod;
            const float bias = b1e[n];
#pragma unroll
            for (int r = 0; r < 8; ++r) {
                const int m = mbase + r + (lhalf << 3);
                hbuf[m * Fc + n] = (__bf16)gelu_erf(acc[j][r] + bias);
            }
        }
    }
    __syncthreads();

    // -------- stage B: out += gate * (h @ w2 + b2) --------
    const __bf16* w2e  = w2t + (size_t)e * Hc * Fc;
    const float*  b2e  = b2 + e * Hc;
    const __bf16* hrow = hbuf + (size_t)(mbase + lmod) * Fc;

    for (int g = 0; g < 4; ++g) {                  // 4 groups of 64 columns
        const int n0 = wn * 256 + g * 64;
        floatx8 acc[4] = {};
        for (int k0 = 0; k0 < Fc; k0 += 32) {
            AFrag a;
            const __bf16* ap = hrow + k0 + (lhalf << 4);
            a.h[0] = *(const bf16x8*)(ap);
            a.h[1] = *(const bf16x8*)(ap + 8);
#pragma unroll
            for (int j = 0; j < 4; ++j) {
                const int n = n0 + j * 16 + lmod;
                const __bf16* bp = w2e + (size_t)n * Fc + k0 + (lhalf << 4);
                bf16x16 b = *(const bf16x16*)bp;
                acc[j] = __builtin_amdgcn_wmma_f32_16x16x32_bf16(
                    false, a.v, false, b, (short)0, acc[j], false, false);
            }
        }
#pragma unroll
        for (int j = 0; j < 4; ++j) {
            const int n = n0 + j * 16 + lmod;
            const float bias = b2e[n];
#pragma unroll
            for (int r = 0; r < 8; ++r) {
                const int m = mbase + r + (lhalf << 3);
                if (s_val[m]) {
                    float v = (acc[j][r] + bias) * s_gw[m];
                    unsafeAtomicAdd(out + (size_t)s_tok[m] * Hc + n, v);
                }
            }
        }
    }
}

// ---------------------------------------------------------------------------
// Host launcher
// ---------------------------------------------------------------------------
extern "C" void kernel_launch(void* const* d_in, const int* in_sizes, int n_in,
                              void* d_out, int out_size, void* d_ws, size_t ws_size,
                              hipStream_t stream) {
    const float* x  = (const float*)d_in[0];
    const float* Wg = (const float*)d_in[1];
    const float* w1 = (const float*)d_in[2];
    const float* b1 = (const float*)d_in[3];
    const float* w2 = (const float*)d_in[4];
    const float* b2 = (const float*)d_in[5];
    float* out = (float*)d_out;

    char* ws = (char*)d_ws;
    size_t off = 0;
    auto carve = [&](size_t bytes) -> void* {
        void* p = ws + off;
        off += (bytes + 255) & ~(size_t)255;
        return p;
    };
    __bf16* xb   = (__bf16*)carve((size_t)NT * Hc * sizeof(__bf16));       // 16.8 MB
    __bf16* w1t  = (__bf16*)carve((size_t)Ec * Fc * Hc * sizeof(__bf16));  // 67.1 MB
    __bf16* w2t  = (__bf16*)carve((size_t)Ec * Hc * Fc * sizeof(__bf16));  // 67.1 MB
    float*  topw = (float*)carve((size_t)NT * KT * sizeof(float));
    int*    tokl = (int*)carve((size_t)Ec * NT * sizeof(int));
    int*    cnt  = (int*)carve((size_t)Ec * sizeof(int));
    (void)ws_size; (void)in_sizes; (void)n_in; (void)out_size;

    const int nOut = NT * Hc;

    // 0) zero out + counters
    moe_init_kernel<<<(nOut + 255) / 256, 256, 0, stream>>>(out, cnt, nOut);
    // 1) x -> bf16
    moe_cvt_x_kernel<<<(nOut + 255) / 256, 256, 0, stream>>>(x, xb, nOut);
    // 2) weight transpose+convert: w1 [E,H,F]->[E,F,H], w2 [E,F,H]->[E,H,F]
    moe_transpose_cvt_kernel<<<dim3(Fc / 32, Hc / 32, Ec), dim3(32, 8), 0, stream>>>(w1, w1t, Hc, Fc);
    moe_transpose_cvt_kernel<<<dim3(Hc / 32, Fc / 32, Ec), dim3(32, 8), 0, stream>>>(w2, w2t, Fc, Hc);
    // 3) gating + routing lists
    moe_gating_kernel<<<NT / 8, 256, 0, stream>>>(x, Wg, topw, tokl, cnt);
    // 4) fused routed FFN (worst-case grid, early exit on device counts)
    moe_ffn_kernel<<<dim3(NT / TM, Ec), 256, 0, stream>>>(xb, w1t, b1, w2t, b2,
                                                          topw, tokl, cnt, out);
}
